// Attention_62818191671511
// MI455X (gfx1250) — matrix operand order repaired and checked by reading, not compiled, not run on hardware
//
#include <hip/hip_runtime.h>

#define S_LEN 2048
#define DIM   4096
#define HD    128
#define NH    32
#define NKV   8
#define KVDIM 1024

typedef __attribute__((ext_vector_type(16))) __bf16        v16bf;
typedef __attribute__((ext_vector_type(8)))  float          v8f;
typedef __attribute__((ext_vector_type(8)))  unsigned short us8;
typedef __attribute__((ext_vector_type(4)))  float          vf4;
typedef __attribute__((ext_vector_type(4)))  unsigned int   u32x4;
typedef __attribute__((ext_vector_type(8)))  int            i32x8;
typedef __attribute__((ext_vector_type(4)))  int            i32x4;

#if __has_builtin(__builtin_amdgcn_tensor_load_to_lds) && \
    __has_builtin(__builtin_amdgcn_s_wait_tensorcnt)
#define USE_TDM 1
#else
#define USE_TDM 0
#endif

union Frag16 {
    v16bf v;
    us8   h[2];
    unsigned short u[16];
};

__device__ __forceinline__ unsigned short f2bf_bits(float f) {
    union { float f; unsigned u; } c; c.f = f;
    unsigned r = c.u + 0x7FFFu + ((c.u >> 16) & 1u);   // round-to-nearest-even
    return (unsigned short)(r >> 16);
}
__device__ __forceinline__ float bf2f(unsigned short b) {
    union { unsigned u; float f; } c; c.u = ((unsigned)b) << 16;
    return c.f;
}
__device__ __forceinline__ v8f wmma_bf16(v16bf a, v16bf b, v8f acc) {
    return __builtin_amdgcn_wmma_f32_16x16x32_bf16(false, a, false, b,
                                                   (short)0, acc, false, false);
}
__device__ __forceinline__ float rowmax16(float v) {
#pragma unroll
    for (int m = 1; m < 16; m <<= 1) v = fmaxf(v, __shfl_xor(v, m, 32));
    return v;
}
__device__ __forceinline__ float rowsum16(float v) {
#pragma unroll
    for (int m = 1; m < 16; m <<= 1) v += __shfl_xor(v, m, 32);
    return v;
}

#if USE_TDM
// 2-D Tensor-Data-Mover load: global (bf16) tile -> LDS.  D# per ISA 08 §8.3/8.4.
// 6-arg builtin variant: (g0 u32x4, g1 i32x8, g2 i32x4, g3 i32x4, i32x8, cpol)
__device__ __forceinline__ void tdm_load_2d(
    unsigned lds_addr, const void* gaddr,
    unsigned tensor_d0, unsigned tensor_d1,
    unsigned tile_d0, unsigned tile_d1, unsigned stride0) {
    const unsigned long long ga = (unsigned long long)gaddr;
    u32x4 g0;
    g0[0] = 1u;                                   // count=1, user descriptor
    g0[1] = lds_addr;                             // LDS byte address
    g0[2] = (unsigned)ga;                         // global addr [31:0]
    g0[3] = (unsigned)(ga >> 32) | (2u << 30);    // addr [56:32] | type=2
    i32x8 g1;
    g1[0] = 0x00010000;                           // data_size = 2 bytes
    g1[1] = (int)((tensor_d0 & 0xFFFFu) << 16);   // tensor_dim0 lo
    g1[2] = (int)((tensor_d0 >> 16) | ((tensor_d1 & 0xFFFFu) << 16));
    g1[3] = (int)((tensor_d1 >> 16) | (tile_d0 << 16));   // tile_dim0
    g1[4] = (int)tile_d1;                         // tile_dim1 (tile_dim2 = 0)
    g1[5] = (int)stride0;                         // tensor_dim0_stride [31:0]
    g1[6] = 0;
    g1[7] = 0;
    i32x4 z4 = {0, 0, 0, 0};
    i32x8 z8 = {0, 0, 0, 0, 0, 0, 0, 0};
    __builtin_amdgcn_tensor_load_to_lds(g0, g1, z4, z4, z8, 0);
}
__device__ __forceinline__ unsigned lds_off(const void* p) {
    return (unsigned)(unsigned long long)p;       // low 32b of flat = LDS offset
}
#endif

// ---------------------------------------------------------------------------
// fp32 -> bf16 streaming convert
// ---------------------------------------------------------------------------
__global__ __launch_bounds__(256) void cvt_bf16_kernel(
    const float* __restrict__ in, unsigned short* __restrict__ out, int n8) {
    int i = blockIdx.x * blockDim.x + threadIdx.x;
    if (i >= n8) return;
    vf4 a = ((const vf4*)in)[2 * i];
    vf4 b = ((const vf4*)in)[2 * i + 1];
    us8 o;
#pragma unroll
    for (int e = 0; e < 4; ++e) o[e]     = f2bf_bits(a[e]);
#pragma unroll
    for (int e = 0; e < 4; ++e) o[4 + e] = f2bf_bits(b[e]);
    ((us8*)out)[i] = o;
}

// ---------------------------------------------------------------------------
// bf16 WMMA GEMM:  Y[M,N] = A[M,D] * W[N,D]^T
//   wave tile 32x64 (2 A-frags x 4 B-frags -> 8 WMMA / K=32 step),
//   8 waves arranged 2(M) x 4(N) -> 64x256 block tile.
// STORE_MODE: 0 = f32 row-major, 1 = bf16 [head][row][d], 2 = bf16 [head][d][row]
// ---------------------------------------------------------------------------
template <int STORE_MODE>
__global__ __launch_bounds__(256) void gemm_wmma(
    const unsigned short* __restrict__ A, const unsigned short* __restrict__ W,
    void* __restrict__ out, int M, int N, int D) {
    const int lane = threadIdx.x & 31;
    const int wave = threadIdx.x >> 5;
    const int half = lane >> 4;
    const int l16  = lane & 15;
    const int row0 = blockIdx.y * 64 + (wave >> 2) * 32;
    const int col0 = blockIdx.x * 256 + (wave & 3) * 64;

    v8f acc[2][4];
#pragma unroll
    for (int i = 0; i < 2; ++i)
#pragma unroll
        for (int j = 0; j < 4; ++j)
#pragma unroll
            for (int e = 0; e < 8; ++e) acc[i][j][e] = 0.f;

    const unsigned short* ap0 = A + (size_t)(row0 + l16) * D;
    const unsigned short* ap1 = A + (size_t)(row0 + 16 + l16) * D;
    const unsigned short* wp[4];
#pragma unroll
    for (int j = 0; j < 4; ++j)
        wp[j] = W + (size_t)(col0 + j * 16 + l16) * D;

    for (int k = 0; k < D; k += 32) {
        Frag16 a0, a1;
        a0.h[0] = *(const us8*)(ap0 + k + half * 8);
        a0.h[1] = *(const us8*)(ap0 + k + 16 + half * 8);
        a1.h[0] = *(const us8*)(ap1 + k + half * 8);
        a1.h[1] = *(const us8*)(ap1 + k + 16 + half * 8);
        __builtin_prefetch(ap0 + k + 256, 0, 1);
        __builtin_prefetch(ap1 + k + 256, 0, 1);
#pragma unroll
        for (int j = 0; j < 4; ++j) {
            Frag16 b;
            const unsigned short* p = wp[j] + k + half * 8;
            b.h[0] = *(const us8*)(p);
            b.h[1] = *(const us8*)(p + 16);
            __builtin_prefetch(p + 256, 0, 1);
            acc[0][j] = wmma_bf16(a0.v, b.v, acc[0][j]);
            acc[1][j] = wmma_bf16(a1.v, b.v, acc[1][j]);
        }
    }

#pragma unroll
    for (int i = 0; i < 2; ++i) {
#pragma unroll
        for (int j = 0; j < 4; ++j) {
            const int col = col0 + j * 16 + l16;
#pragma unroll
            for (int r = 0; r < 8; ++r) {
                const int row = row0 + 16 * i + r + half * 8;
                const float val = acc[i][j][r];
                if constexpr (STORE_MODE == 0) {
                    ((float*)out)[(size_t)row * N + col] = val;
                } else if constexpr (STORE_MODE == 1) {
                    const int hh = col >> 7, dd = col & 127;
                    ((unsigned short*)out)[((size_t)hh * M + row) * HD + dd] =
                        f2bf_bits(val);
                } else {   // transposed head-major (for V)
                    const int hh = col >> 7, dd = col & 127;
                    ((unsigned short*)out)[((size_t)hh * HD + dd) * M + row] =
                        f2bf_bits(val);
                }
            }
        }
    }
}

// ---------------------------------------------------------------------------
// In-place RoPE on head-major bf16 [H][S][128]
// ---------------------------------------------------------------------------
__global__ __launch_bounds__(256) void rope_kernel(
    unsigned short* __restrict__ q, const float* __restrict__ cosb,
    const float* __restrict__ sinb, int H) {
    int idx = blockIdx.x * blockDim.x + threadIdx.x;
    if (idx >= H * S_LEN * (HD / 2)) return;
    const int i = idx & 63;
    const int s = (idx >> 6) & (S_LEN - 1);
    const int h = idx / (S_LEN * 64);
    const size_t base = ((size_t)h * S_LEN + s) * HD + 2 * i;
    const float v1 = bf2f(q[base]);
    const float v2 = bf2f(q[base + 1]);
    const float c  = cosb[s * 64 + i];
    const float sn = sinb[s * 64 + i];
    q[base]     = f2bf_bits(v1 * c - v2 * sn);
    q[base + 1] = f2bf_bits(v1 * sn + v2 * c);
}

// ---------------------------------------------------------------------------
// Flash attention.  K tile [key][d] and V tile [d][key] are DMA'd to LDS by
// the Tensor Data Mover, double-buffered (tile i+1 in flight during compute
// on tile i).  V comes pre-transposed from global ([h][d][s]).
// ---------------------------------------------------------------------------
__global__ __launch_bounds__(256) void attn_kernel(
    const unsigned short* __restrict__ qh, const unsigned short* __restrict__ kh,
    const unsigned short* __restrict__ vhT, unsigned short* __restrict__ attnb) {
    __shared__ alignas(16) unsigned short kt[2][32][HD];    // K tile [key][d]
    __shared__ alignas(16) unsigned short vt[2][HD][32];    // V tile [d][key]
    __shared__ alignas(16) unsigned short pb[8][16][32];    // probs per wave

    const int lane = threadIdx.x & 31;
    const int wave = threadIdx.x >> 5;
    const int half = lane >> 4;
    const int l16  = lane & 15;
    const int h    = blockIdx.y;
    const int hk   = h >> 2;                     // GQA 32 -> 8 kv heads
    const int m0   = blockIdx.x * 128 + wave * 16;

    Frag16 qa[4];
    {
        const unsigned short* qp = qh + ((size_t)h * S_LEN + (m0 + l16)) * HD;
#pragma unroll
        for (int c = 0; c < 4; ++c) {
            qa[c].h[0] = *(const us8*)(qp + 32 * c + half * 16);
            qa[c].h[1] = *(const us8*)(qp + 32 * c + half * 16 + 8);
        }
    }

    v8f   o[8];
    float mrun[8], lrun[8];
#pragma unroll
    for (int c = 0; c < 8; ++c)
#pragma unroll
        for (int e = 0; e < 8; ++e) o[c][e] = 0.f;
#pragma unroll
    for (int r = 0; r < 8; ++r) { mrun[r] = -3.0e38f; lrun[r] = 0.f; }

    const float scale = 0.08838834764831845f;    // 128^-0.5
    const int nIter = S_LEN / 32;

#if USE_TDM
    // Issue DMA for tile 0 (wave 0 only; TDM ignores EXEC, branch is uniform).
    if (wave == 0) {
        tdm_load_2d(lds_off(kt[0]), kh + ((size_t)hk * S_LEN) * HD,
                    HD, S_LEN, HD, 32, HD);
        tdm_load_2d(lds_off(vt[0]), vhT + ((size_t)hk * HD) * S_LEN,
                    S_LEN, HD, 32, HD, S_LEN);
    }
#endif

    for (int it = 0; it < nIter; ++it) {
        const int buf = it & 1;
#if USE_TDM
        if (wave == 0) {
            if (it + 1 < nIter) {
                const int kb = (it + 1) * 32;
                tdm_load_2d(lds_off(kt[buf ^ 1]),
                            kh + (((size_t)hk * S_LEN + kb)) * HD,
                            HD, S_LEN, HD, 32, HD);
                tdm_load_2d(lds_off(vt[buf ^ 1]),
                            vhT + ((size_t)hk * HD) * S_LEN + kb,
                            S_LEN, HD, 32, HD, S_LEN);
                __builtin_amdgcn_s_wait_tensorcnt(2);  // tile `it` complete
            } else {
                __builtin_amdgcn_s_wait_tensorcnt(0);
            }
        }
#else
        {   // fallback: cooperative vector copy
            const int kb = it * 32;
            const int r  = threadIdx.x >> 3;
            const int cc = (threadIdx.x & 7) * 16;
            const unsigned short* kp = kh + ((size_t)hk * S_LEN + kb + r) * HD + cc;
            *(us8*)&kt[buf][r][cc]     = *(const us8*)kp;
            *(us8*)&kt[buf][r][cc + 8] = *(const us8*)(kp + 8);
            const int d  = threadIdx.x >> 1;
            const int c2 = (threadIdx.x & 1) * 16;
            const unsigned short* vp =
                vhT + ((size_t)hk * HD + d) * S_LEN + kb + c2;
            *(us8*)&vt[buf][d][c2]     = *(const us8*)vp;
            *(us8*)&vt[buf][d][c2 + 8] = *(const us8*)(vp + 8);
        }
#endif
        __syncthreads();

        // scores: Q(16x128) @ K_tile^T -> two 16x16 f32 tiles
        v8f s0, s1;
#pragma unroll
        for (int e = 0; e < 8; ++e) { s0[e] = 0.f; s1[e] = 0.f; }
#pragma unroll
        for (int c = 0; c < 4; ++c) {
            Frag16 b0, b1;
            const unsigned short* p0 = &kt[buf][l16][32 * c + half * 16];
            b0.h[0] = *(const us8*)p0;
            b0.h[1] = *(const us8*)(p0 + 8);
            const unsigned short* p1 = &kt[buf][16 + l16][32 * c + half * 16];
            b1.h[0] = *(const us8*)p1;
            b1.h[1] = *(const us8*)(p1 + 8);
            s0 = wmma_bf16(qa[c].v, b0.v, s0);
            s1 = wmma_bf16(qa[c].v, b1.v, s1);
        }

        // online softmax across this 32-key block
        float alpha[8];
#pragma unroll
        for (int r = 0; r < 8; ++r) {
            const float x0 = s0[r] * scale;
            const float x1 = s1[r] * scale;
            const float mx   = rowmax16(fmaxf(x0, x1));
            const float mnew = fmaxf(mrun[r], mx);
            const float a    = __expf(mrun[r] - mnew);
            const float p0   = __expf(x0 - mnew);
            const float p1   = __expf(x1 - mnew);
            lrun[r] = lrun[r] * a + rowsum16(p0 + p1);
            mrun[r] = mnew;
            alpha[r] = a;
            const int rr = r + half * 8;
            pb[wave][rr][l16]      = f2bf_bits(p0);
            pb[wave][rr][16 + l16] = f2bf_bits(p1);
        }

#pragma unroll
        for (int c = 0; c < 8; ++c)
#pragma unroll
            for (int r = 0; r < 8; ++r) o[c][r] *= alpha[r];

        // PV: probs(16x32) @ V_tile(32x128)
        Frag16 pa;
        pa.h[0] = *(const us8*)&pb[wave][l16][half * 16];
        pa.h[1] = *(const us8*)&pb[wave][l16][half * 16 + 8];
#pragma unroll
        for (int c = 0; c < 8; ++c) {
            Frag16 vb;
            const unsigned short* p = &vt[buf][16 * c + l16][half * 16];
            vb.h[0] = *(const us8*)p;
            vb.h[1] = *(const us8*)(p + 8);
            o[c] = wmma_bf16(pa.v, vb.v, o[c]);
        }
        __syncthreads();
    }

#pragma unroll
    for (int r = 0; r < 8; ++r) {
        const float inv = 1.0f / lrun[r];
        const int row = m0 + r + half * 8;
#pragma unroll
        for (int c = 0; c < 8; ++c)
            attnb[(size_t)row * DIM + h * HD + 16 * c + l16] =
                f2bf_bits(o[c][r] * inv);
    }
}

// ---------------------------------------------------------------------------
extern "C" void kernel_launch(void* const* d_in, const int* in_sizes, int n_in,
                              void* d_out, int out_size, void* d_ws, size_t ws_size,
                              hipStream_t stream) {
    (void)in_sizes; (void)n_in; (void)out_size; (void)ws_size;
    const float* x    = (const float*)d_in[0];
    const float* cosb = (const float*)d_in[1];
    const float* sinb = (const float*)d_in[2];
    const float* wq   = (const float*)d_in[3];
    const float* wk   = (const float*)d_in[4];
    const float* wv   = (const float*)d_in[5];
    const float* wo   = (const float*)d_in[6];
    float* out = (float*)d_out;

    unsigned short* p   = (unsigned short*)d_ws;
    unsigned short* xb  = p; p += (size_t)S_LEN * DIM;
    unsigned short* wqb = p; p += (size_t)DIM * DIM;
    unsigned short* wkb = p; p += (size_t)KVDIM * DIM;
    unsigned short* wvb = p; p += (size_t)KVDIM * DIM;
    unsigned short* wob = p; p += (size_t)DIM * DIM;
    unsigned short* qhb = p; p += (size_t)NH  * S_LEN * HD;
    unsigned short* khb = p; p += (size_t)NKV * S_LEN * HD;
    unsigned short* vhT = p; p += (size_t)NKV * S_LEN * HD;   // [h][d][s]
    unsigned short* attnb = p;

    auto cvt = [&](const float* src, unsigned short* dst, size_t n) {
        int n8 = (int)(n / 8);
        cvt_bf16_kernel<<<dim3((n8 + 255) / 256), dim3(256), 0, stream>>>(src, dst, n8);
    };
    cvt(x,  xb,  (size_t)S_LEN * DIM);
    cvt(wq, wqb, (size_t)DIM * DIM);
    cvt(wk, wkb, (size_t)KVDIM * DIM);
    cvt(wv, wvb, (size_t)KVDIM * DIM);
    cvt(wo, wob, (size_t)DIM * DIM);

    // QKV projections (Q,K head-major; V transposed head-major for TDM tiles)
    gemm_wmma<1><<<dim3(DIM / 256,   S_LEN / 64), dim3(256), 0, stream>>>(
        xb, wqb, qhb, S_LEN, DIM, DIM);
    gemm_wmma<1><<<dim3(KVDIM / 256, S_LEN / 64), dim3(256), 0, stream>>>(
        xb, wkb, khb, S_LEN, KVDIM, DIM);
    gemm_wmma<2><<<dim3(KVDIM / 256, S_LEN / 64), dim3(256), 0, stream>>>(
        xb, wvb, vhT, S_LEN, KVDIM, DIM);

    rope_kernel<<<dim3(NH  * S_LEN * 64 / 256), dim3(256), 0, stream>>>(qhb, cosb, sinb, NH);
    rope_kernel<<<dim3(NKV * S_LEN * 64 / 256), dim3(256), 0, stream>>>(khb, cosb, sinb, NKV);

    attn_kernel<<<dim3(S_LEN / 128, NH), dim3(256), 0, stream>>>(qhb, khb, vhT, attnb);

    gemm_wmma<0><<<dim3(DIM / 256, S_LEN / 64), dim3(256), 0, stream>>>(
        attnb, wob, out, S_LEN, DIM, DIM);
}
